// CorticalMap_77060303225234
// MI455X (gfx1250) — matrix operand order repaired and checked by reading, not compiled, not run on hardware
//
#include <hip/hip_runtime.h>
#include <hip/hip_bf16.h>
#include <math.h>

typedef __attribute__((ext_vector_type(16))) _Float16 v16h;
typedef __attribute__((ext_vector_type(8)))  float    v8f;

#define S_   128
#define KA_  25
#define C_   2
#define KL_  25
#define FA_  (C_ * KA_ * KA_)   // 1250
#define FL_  (KL_ * KL_)        // 625
#define NL_  (S_ * S_)          // 16384
#define INW_ (S_ + KA_ - 1)     // 152
#define PI_F 3.14159265358979323846f
#define HOMEO_ 0.04f

// ---------------------------------------------------------------------------
// Kernel 0: compute AFF_ENV (625) and LRI_ENV (625) envelopes in workspace.
// AFF_ENV = cos(d*pi/25)^2 * (d<12.5); its max is exactly 1 (center), so no
// normalization. LRI_ENV needs a max-normalize (center is zeroed by the
// inhibition term), done with an LDS tree reduction.
// ---------------------------------------------------------------------------
__global__ void __launch_bounds__(1024) init_envs(float* __restrict__ aff_env,
                                                  float* __restrict__ lri_env) {
    __shared__ float red[1024];
    int t = threadIdx.x;
    float ae = 0.f, le = 0.f;
    if (t < FL_) {
        int ky = t / KL_, kx = t - ky * KL_;
        float gy = (float)ky - 12.f, gx = (float)kx - 12.f;
        float d  = sqrtf(gy * gy + gx * gx);
        float c25 = cosf(d * PI_F / 25.f);
        float circ12 = (d < 12.5f) ? 1.f : 0.f;
        ae = c25 * c25 * circ12;                       // AFF_ENV (already max=1)
        float c9  = cosf(d * PI_F / 9.f);
        float inh = c9 * c9 * ((d < 4.5f) ? 1.f : 0.f);
        le = c25 * c25 * (1.f - inh) * circ12;
    }
    red[t] = le;
    __syncthreads();
    for (int s = 512; s > 0; s >>= 1) {
        if (t < s) red[t] = fmaxf(red[t], red[t + s]);
        __syncthreads();
    }
    float m = red[0];
    if (t < FL_) {
        aff_env[t] = ae;
        lri_env[t] = le / m;
    }
}

// ---------------------------------------------------------------------------
// Kernel 1: tiles[l, f] = x[c, row+ky, col+kx] * AFF_ENV[kk]
// l = row*128 + col, f = c*625 + ky*25 + kx (torch unfold ordering).
// One block per location; threads stride over f (coalesced over kx).
// ---------------------------------------------------------------------------
__global__ void __launch_bounds__(256) tiles_kernel(const float* __restrict__ x,
                                                    const float* __restrict__ aff_env,
                                                    float* __restrict__ tiles) {
    int l   = blockIdx.x;
    int row = l >> 7, col = l & 127;
    for (int f = threadIdx.x; f < FA_; f += blockDim.x) {
        int c  = f / (KA_ * KA_);
        int kk = f - c * (KA_ * KA_);
        int ky = kk / KA_, kx = kk - ky * KA_;
        float v = x[(c * INW_ + row + ky) * INW_ + col + kx] * aff_env[kk];
        tiles[(size_t)l * FA_ + f] = v;
    }
}

// ---------------------------------------------------------------------------
// Kernel 2: raw_aff[l] = dot(tiles[l,:], relu(rfs[l,:])) via WMMA diag trick.
// One BLOCK (8 waves) per group of 16 locations; the 39 full K-chunks of 32
// are split 5 per wave (wave 7: 4 chunks + exact-f32 tail for f=1248,1249).
//   A[m,k] = tiles[l0+m, f0+k]      (16x32 f16)
//   B[k,n] = relu(rfs[l0+n, f0+k])  (32x16 f16)
//   D += A x B  => diag(D)[m] accumulates the per-location dots (f32 accum).
// Partial diagonals are combined through LDS in a fixed (deterministic) order.
// Per-lane K order is contiguous (K = 16h + s), the same in A and B.
// ---------------------------------------------------------------------------
__global__ void __launch_bounds__(256) rawaff_wmma(const float* __restrict__ tiles,
                                                   const float* __restrict__ rfs,
                                                   float* __restrict__ raw_aff_out) {
    __shared__ float part[8 * 16];
    int w    = threadIdx.x >> 5;    // wave in block: 0..7 (uniform per wave)
    int lane = threadIdx.x & 31;
    int l0   = blockIdx.x << 4;     // 16-location group
    int h    = lane >> 4;           // wave half
    int r    = lane & 15;           // row (A: M, B: N)

    const float* trow = tiles + (size_t)(l0 + r) * FA_;
    const float* rrow = rfs   + (size_t)(l0 + r) * FA_;

    const int aoff0 = 16 * h;       // A halves 0..7  -> K = 16h + j
    const int aoff1 = 16 * h + 8;   // A halves 8..15 -> K = 16h + 8 + j
    const int boff  = 16 * h;       // B halves j     -> K = 16h + j

    v8f c = {0.f, 0.f, 0.f, 0.f, 0.f, 0.f, 0.f, 0.f};

    int cb = 5 * w;
    int ce = (w == 7) ? 39 : (cb + 5);      // wave 7 owns 4 chunks + tail
    for (int ch = cb; ch < ce; ++ch) {
        int f0 = ch * 32;
        // Stream hint for the rows we are marching through (gfx1250 prefetch).
        __builtin_prefetch(trow + f0 + 64, 0, 0);
        __builtin_prefetch(rrow + f0 + 64, 0, 0);
        v16h a, b;
#pragma unroll
        for (int j = 0; j < 8; ++j) {
            a[j]     = (_Float16)trow[f0 + aoff0 + j];
            a[8 + j] = (_Float16)trow[f0 + aoff1 + j];
        }
#pragma unroll
        for (int j = 0; j < 16; ++j) {
            float bv = rrow[f0 + boff + j];
            b[j] = (_Float16)(bv > 0.f ? bv : 0.f);
        }
        c = __builtin_amdgcn_wmma_f32_16x16x32_f16(false, a, false, b,
                                                   (short)0, c, false, false);
    }

    // Diagonal extraction: lanes 0-7 hold diag m=lane in c[lane];
    // lanes 24-31 hold diag m=lane-16 in c[lane-24].
    bool valid = (lane < 8) || (lane >= 24);
    int  m = (lane < 8) ? lane : (lane - 16);
    int  i = m & 7;
    float d = c[0];
    if (i == 1) d = c[1];
    if (i == 2) d = c[2];
    if (i == 3) d = c[3];
    if (i == 4) d = c[4];
    if (i == 5) d = c[5];
    if (i == 6) d = c[6];
    if (i == 7) d = c[7];

    if (w == 7 && valid) {
        // Exact f32 tail: remaining features f = 1248, 1249 for row l0+m.
        const float* tr = tiles + (size_t)(l0 + m) * FA_;
        const float* rr = rfs   + (size_t)(l0 + m) * FA_;
        float r0 = rr[FA_ - 2]; r0 = r0 > 0.f ? r0 : 0.f;
        float r1 = rr[FA_ - 1]; r1 = r1 > 0.f ? r1 : 0.f;
        d += tr[FA_ - 2] * r0 + tr[FA_ - 1] * r1;
    }
    if (valid) part[w * 16 + m] = d;
    __syncthreads();

    if (threadIdx.x < 16) {
        float s = 0.f;
        for (int ww = 0; ww < 8; ++ww) s += part[ww * 16 + threadIdx.x];
        raw_aff_out[l0 + threadIdx.x] = s;
    }
}

// ---------------------------------------------------------------------------
// Kernel 3: lateral inhibition + final activation.
// aff = raw_aff - adathresh; lat0 = relu(aff) (zero-padded OOB);
// lat_neg[l] = sum_j LRI[j] * lat0_pad(l,j) * relu(lw[l,j]);
// lat[l] = tanh(relu(lat0[l] - lat_neg[l] + aff[l])).
// One wave per location; shuffle tree reduction.
// ---------------------------------------------------------------------------
__global__ void __launch_bounds__(256) lat_kernel(const float* __restrict__ raw_aff,
                                                  const float* __restrict__ ada,
                                                  const float* __restrict__ lw,
                                                  const float* __restrict__ lri_env,
                                                  float* __restrict__ lat_out) {
    int lane = threadIdx.x & 31;
    int l    = blockIdx.x * (blockDim.x >> 5) + (threadIdx.x >> 5);
    int row  = l >> 7, col = l & 127;
    const float* lwrow = lw + (size_t)l * FL_;

    float p = 0.f;
    for (int j = lane; j < FL_; j += 32) {
        int ky = j / KL_, kx = j - ky * KL_;
        int y = row + ky - 12, xx = col + kx - 12;
        float v = 0.f;
        if (y >= 0 && y < S_ && xx >= 0 && xx < S_) {
            int g = y * S_ + xx;
            float a = raw_aff[g] - ada[g];
            v = a > 0.f ? a : 0.f;
        }
        float wv = lwrow[j];
        wv = wv > 0.f ? wv : 0.f;
        p += lri_env[j] * v * wv;
    }
    p += __shfl_xor(p, 16);
    p += __shfl_xor(p, 8);
    p += __shfl_xor(p, 4);
    p += __shfl_xor(p, 2);
    p += __shfl_xor(p, 1);

    if (lane == 0) {
        float affl = raw_aff[l] - ada[l];
        float lat0 = affl > 0.f ? affl : 0.f;
        float t = lat0 - p + affl;          // STRENGTH == 1
        t = t > 0.f ? t : 0.f;
        lat_out[l] = tanhf(t);
    }
}

// ---------------------------------------------------------------------------
// Kernel 4: per-block partial sums of
//   sum_l lat[l] * sum_j LRI[j] * lat_pad(l,j; pad=HOMEO) * relu(lw[l,j])
// One wave per location, 8 locations per block -> one partial per block.
// ---------------------------------------------------------------------------
__global__ void __launch_bounds__(256) corr_kernel(const float* __restrict__ lat,
                                                   const float* __restrict__ lw,
                                                   const float* __restrict__ lri_env,
                                                   float* __restrict__ partials) {
    __shared__ float acc[8];
    int wib  = threadIdx.x >> 5;
    int lane = threadIdx.x & 31;
    int l    = blockIdx.x * 8 + wib;
    int row  = l >> 7, col = l & 127;
    const float* lwrow = lw + (size_t)l * FL_;

    float p = 0.f;
    for (int j = lane; j < FL_; j += 32) {
        int ky = j / KL_, kx = j - ky * KL_;
        int y = row + ky - 12, xx = col + kx - 12;
        float v = (y >= 0 && y < S_ && xx >= 0 && xx < S_) ? lat[y * S_ + xx] : HOMEO_;
        float wv = lwrow[j];
        wv = wv > 0.f ? wv : 0.f;
        p += lri_env[j] * v * wv;
    }
    p += __shfl_xor(p, 16);
    p += __shfl_xor(p, 8);
    p += __shfl_xor(p, 4);
    p += __shfl_xor(p, 2);
    p += __shfl_xor(p, 1);
    if (lane == 0) acc[wib] = p * lat[l];
    __syncthreads();
    if (threadIdx.x == 0) {
        float s = 0.f;
        for (int i = 0; i < 8; ++i) s += acc[i];
        partials[blockIdx.x] = s;
    }
}

// Kernel 5: deterministic final reduction of 2048 partials -> scalar.
__global__ void __launch_bounds__(256) finalsum(const float* __restrict__ partials,
                                                float* __restrict__ out_scalar) {
    __shared__ float red[256];
    float s = 0.f;
    for (int i = threadIdx.x; i < 2048; i += 256) s += partials[i];
    red[threadIdx.x] = s;
    __syncthreads();
    for (int st = 128; st > 0; st >>= 1) {
        if (threadIdx.x < st) red[threadIdx.x] += red[threadIdx.x + st];
        __syncthreads();
    }
    if (threadIdx.x == 0) out_scalar[0] = red[0];
}

// ---------------------------------------------------------------------------
extern "C" void kernel_launch(void* const* d_in, const int* in_sizes, int n_in,
                              void* d_out, int out_size, void* d_ws, size_t ws_size,
                              hipStream_t stream) {
    (void)in_sizes; (void)n_in; (void)out_size; (void)ws_size;
    const float* x   = (const float*)d_in[0];   // [1,2,152,152]
    const float* rfs = (const float*)d_in[1];   // [16384,1250,1]
    const float* lw  = (const float*)d_in[2];   // [16384,625,1]
    const float* ada = (const float*)d_in[3];   // [1,1,128,128]

    float* out      = (float*)d_out;
    float* raw_aff  = out;                  // 16384
    float* lat      = out + NL_;            // 16384
    float* corr     = out + 2 * NL_;        // 1
    float* tiles    = out + 2 * NL_ + 1;    // 16384*1250

    float* ws       = (float*)d_ws;
    float* aff_env  = ws;                   // 625
    float* lri_env  = ws + FL_;             // 625
    float* partials = ws + 2 * FL_;         // 2048

    init_envs  <<<1,    1024, 0, stream>>>(aff_env, lri_env);
    tiles_kernel<<<NL_, 256,  0, stream>>>(x, aff_env, tiles);
    rawaff_wmma<<<NL_ / 16, 256, 0, stream>>>(tiles, rfs, raw_aff); // 1024 blocks, 8192 waves
    lat_kernel <<<2048, 256,  0, stream>>>(raw_aff, ada, lw, lri_env, lat);
    corr_kernel<<<2048, 256,  0, stream>>>(lat, lw, lri_env, partials);
    finalsum   <<<1,    256,  0, stream>>>(partials, corr);
}
